// M_1803886265414
// MI455X (gfx1250) — matrix operand order repaired and checked by reading, not compiled, not run on hardware
//
#include <hip/hip_runtime.h>
#include <hip/hip_bf16.h>

typedef __attribute__((ext_vector_type(16))) _Float16 v16h;
typedef __attribute__((ext_vector_type(4)))  _Float16 v4h;
typedef __attribute__((ext_vector_type(8)))  float    v8f;

#define K_DIM   384
#define N_DIM   1536
#define BM      64         // rows (tokens) per workgroup
#define KSTEPS  12         // 384 / 32

// ---------------------------------------------------------------------------
// Kernel 0: convert W (1536x384 f32 row-major) into a fragment-swizzled f16
// buffer: Whf[((nt*12 + ks)*32 + lane)*16 + e]
//            = W[nt*16 + (lane&15)][ks*32 + (lane>>4)*8 + (e&7) + (e>>3)*16]
// Matches CDNA5 16-bit B (KxN = 32x16) fragment layout; a wave's B-fragment
// load is 1 KB contiguous. Strides: 512 halves per ks, 6144 halves per nt.
// ---------------------------------------------------------------------------
__global__ void wcvt_kernel(const float* __restrict__ W,
                            _Float16* __restrict__ Whf) {
  int tid  = blockIdx.x * 256 + threadIdx.x;   // 96*12*32*16 = 589824 threads
  int e    = tid & 15;
  int lane = (tid >> 4) & 31;
  int rest = tid >> 9;
  int ks   = rest % 12;
  int nt   = rest / 12;
  int n    = nt * 16 + (lane & 15);
  int k    = ks * 32 + (lane >> 4) * 8 + (e & 7) + (e >> 3) * 16;
  Whf[tid] = (_Float16)W[n * K_DIM + k];
}

// ---------------------------------------------------------------------------
// Kernel 1: fused LayerNorm(384) -> Linear(384->1536) -> exact-erf GELU.
// 784 workgroups x 256 threads (8 wave32); each WG owns 64 tokens.
// xn is stored in LDS pre-swizzled into WMMA A-fragment order so a fragment
// is one contiguous 32B read per lane (two ds_load_b128, no register copies).
// A-fragment LDS layout: lds_a[((mt*12+ks)*32 + lane)*16 + e]
//   lane = (m&15) + 16*kh ; e<8 -> K = ks*32 + 8*kh + e
//                           e>=8 -> K = ks*32 + 16 + 8*kh + (e-8)
// ---------------------------------------------------------------------------
__global__ __launch_bounds__(256, 4) void ln_gemm_gelu_kernel(
    const float* __restrict__ x, const float* __restrict__ gamma,
    const float* __restrict__ beta, const _Float16* __restrict__ Whf,
    const float* __restrict__ bias, float* __restrict__ out) {
  __shared__ __align__(16) _Float16 lds_a[4 * KSTEPS * 32 * 16];  // 49152 B

  const int tid = threadIdx.x;

  // ---------------- Phase 1: LayerNorm (4 threads per row) ----------------
  {
    const int  row  = tid >> 2;       // 0..63
    const int  q    = tid & 3;        // quarter of the 384-wide row
    const long grow = (long)blockIdx.x * BM + row;
    const float4* xp = (const float4*)(x + grow * K_DIM + q * 96);

    float4 vals[24];
    float s = 0.f, sq = 0.f;
#pragma unroll
    for (int i = 0; i < 24; ++i) {
      float4 v = xp[i];
      vals[i] = v;
      s  += v.x + v.y + v.z + v.w;
      sq += v.x * v.x + v.y * v.y + v.z * v.z + v.w * v.w;
    }
    // 4 cooperating threads are lanes {t, t^1, t^2, t^3} of one wave32
    s  += __shfl_xor(s, 1);  s  += __shfl_xor(s, 2);
    sq += __shfl_xor(sq, 1); sq += __shfl_xor(sq, 2);

    const float mean = s * (1.0f / 384.0f);
    const float rstd = rsqrtf(sq * (1.0f / 384.0f) - mean * mean + 1e-5f);

    const int mt  = row >> 4;
    const int r15 = row & 15;
#pragma unroll
    for (int i = 0; i < 24; ++i) {
      const int k  = q * 96 + i * 4;
      const int ks = k >> 5;
      const int ko = k & 31;
      const int kh = (ko >> 3) & 1;                  // which lane half-group
      const int e0 = (ko & 7) + ((ko >> 4) & 1) * 8; // element within fragment
      const int off = (mt * KSTEPS + ks) * 512 + (r15 + 16 * kh) * 16 + e0;

      float4 g4 = *(const float4*)(gamma + k);
      float4 b4 = *(const float4*)(beta + k);
      v4h h;
      h[0] = (_Float16)((vals[i].x - mean) * rstd * g4.x + b4.x);
      h[1] = (_Float16)((vals[i].y - mean) * rstd * g4.y + b4.y);
      h[2] = (_Float16)((vals[i].z - mean) * rstd * g4.z + b4.z);
      h[3] = (_Float16)((vals[i].w - mean) * rstd * g4.w + b4.w);
      *(v4h*)&lds_a[off] = h;
    }
  }
  __syncthreads();

  // ---------------- Phase 2: GEMM via v_wmma_f32_16x16x32_f16 -------------
  const int  wave = tid >> 5;
  const int  lane = tid & 31;
  const int  m16  = lane & 15;
  const int  kh   = lane >> 4;
  const long orow0 = (long)blockIdx.x * BM;

  // Single LDS base; A loads use compile-time DS byte offsets:
  //   off = mt*12288 + ks*1024   (max 48128 < 65536)
  const _Float16* abase = &lds_a[lane * 16];

#pragma clang loop unroll(disable)
  for (int it = 0; it < 6; ++it) {          // 96 N-tiles = 6 * 8 waves * 2
    // Compiler barrier: A-fragment LDS loads are it-invariant; without this
    // LICM hoists all 48 of them (384 VGPRs) out of the loop and spills to
    // scratch. Re-reading LDS each iteration is the intended (cheap) path.
    asm volatile("" ::: "memory");

    const int nt0 = it * 16 + wave * 2;
    // Single global base; B loads use immediate offsets:
    //   off = j*12288 + ks*1024 bytes     (max 23552)
    const _Float16* bp = Whf + (long)nt0 * (KSTEPS * 512) + lane * 16;

    v8f acc[4][2];
#pragma unroll
    for (int mt = 0; mt < 4; ++mt)
#pragma unroll
      for (int j = 0; j < 2; ++j)
        acc[mt][j] = (v8f){0.f, 0.f, 0.f, 0.f, 0.f, 0.f, 0.f, 0.f};

    // Double-buffer B (global/L2 latency); A comes fresh from LDS each step.
    v16h bcur[2], bnxt[2];
#pragma unroll
    for (int j = 0; j < 2; ++j)
      bcur[j] = *(const v16h*)(bp + j * (KSTEPS * 512));

#pragma unroll
    for (int ks = 0; ks < KSTEPS; ++ks) {
      if (ks + 1 < KSTEPS) {
#pragma unroll
        for (int j = 0; j < 2; ++j)
          bnxt[j] = *(const v16h*)(bp + j * (KSTEPS * 512) + (ks + 1) * 512);
      }
      v16h a[4];
#pragma unroll
      for (int mt = 0; mt < 4; ++mt)
        a[mt] = *(const v16h*)(abase + mt * (KSTEPS * 512) + ks * 512);
#pragma unroll
      for (int j = 0; j < 2; ++j)
#pragma unroll
        for (int mt = 0; mt < 4; ++mt)
          acc[mt][j] = __builtin_amdgcn_wmma_f32_16x16x32_f16(
              false, a[mt], false, bcur[j], (short)0, acc[mt][j],
              false, false);
#pragma unroll
      for (int j = 0; j < 2; ++j)
        bcur[j] = bnxt[j];    // SSA rename under full unroll, no real moves
    }

    // ------------- Epilogue: bias + exact GELU + store f32 ---------------
#pragma unroll
    for (int j = 0; j < 2; ++j) {
      const int   col = (nt0 + j) * 16 + m16;       // C/D: N = lane&15
      const float bv  = bias[col];
      // single base; stores at immediate offsets (mt*16+v)*6144 bytes
      float* op = out + (orow0 + kh * 8) * N_DIM + col;
#pragma unroll
      for (int mt = 0; mt < 4; ++mt) {
#pragma unroll
        for (int v = 0; v < 8; ++v) {               // C/D: M = v + 8*kh
          const float y = acc[mt][j][v] + bv;
          const float g = 0.5f * y * (1.0f + erff(y * 0.70710678118654752f));
          op[(mt * 16 + v) * N_DIM] = g;
        }
      }
    }
  }
}

extern "C" void kernel_launch(void* const* d_in, const int* in_sizes, int n_in,
                              void* d_out, int out_size, void* d_ws, size_t ws_size,
                              hipStream_t stream) {
  (void)in_sizes; (void)n_in; (void)out_size; (void)ws_size;
  const float* x     = (const float*)d_in[0];
  const float* gamma = (const float*)d_in[1];
  const float* beta  = (const float*)d_in[2];
  const float* W     = (const float*)d_in[3];
  const float* b     = (const float*)d_in[4];
  float* out = (float*)d_out;
  _Float16* Whf = (_Float16*)d_ws;   // 1536*384*2 = 1,179,648 bytes

  // 96*12*32*16 = 589824 elements, 256 threads/block -> 2304 blocks
  wcvt_kernel<<<2304, 256, 0, stream>>>(W, Whf);
  // 50176 tokens / 64 per WG = 784 workgroups
  ln_gemm_gelu_kernel<<<784, 256, 0, stream>>>(x, gamma, beta, Whf, b, out);
}
